// nri_rec_encoder_32049045962802
// MI455X (gfx1250) — matrix operand order repaired and checked by reading, compiled, not run on hardware
//
#include <hip/hip_runtime.h>
#include <hip/hip_bf16.h>

// ---------------------------------------------------------------------------
// Problem constants (from the reference)
// ---------------------------------------------------------------------------
#define BATCH   128
#define TSTEPS  1024
#define NNODES  29
#define D_IN    6
#define HDIM    256
#define NEDGES  56            // 2*(N-1)
#define D_COMB  262           // D_IN + H
#define KPAD    288           // D_COMB padded to 9 k-tiles of 32
#define KT_G    9             // k-tiles for gate GEMM
#define NT_G    64            // n-tiles for gate GEMM (4H = 1024)
#define KT_M    16            // k-tiles for mlp1 (K=512)
#define NT_M    16            // n-tiles for mlp1 (N=256)
#define KT_F    8             // k-tiles for fc (K=256)
#define NT_F    2             // n-tiles for fc (N=32)
#define D_OUT   32
#define NTHREADS 512          // 16 waves: small per-wave tile

// fragment strides in bf16 halves
#define FRAG_HALVES   (32 * 16)                   // one 32x16 B tile = 512 halves
#define KSLAB_G       (NT_G * FRAG_HALVES)        // 32768 halves per gate k-slab
#define GATE_OFF      (16 * FRAG_HALVES)          // 8192 halves between gate n-tiles

typedef __attribute__((ext_vector_type(16))) __bf16 v16bf;
typedef __attribute__((ext_vector_type(8)))  float  v8f;

// ---------------------------------------------------------------------------
// bf16 helpers (bit-level, no native conversion dependence)
// ---------------------------------------------------------------------------
__device__ __host__ inline __bf16 f2bf(float f) {
    unsigned u;
    __builtin_memcpy(&u, &f, 4);
    u += 0x7fffu + ((u >> 16) & 1u);        // round-to-nearest-even
    unsigned short s = (unsigned short)(u >> 16);
    __bf16 r;
    __builtin_memcpy(&r, &s, 2);
    return r;
}
__device__ inline float bf2f(__bf16 b) {
    unsigned short s;
    __builtin_memcpy(&s, &b, 2);
    unsigned u = ((unsigned)s) << 16;
    float f;
    __builtin_memcpy(&f, &u, 4);
    return f;
}
// straight-line activations: one v_exp_f32 each, no EXEC-divergent libm paths
__device__ inline float sigm(float x)      { return __frcp_rn(1.0f + __expf(-x)); }
__device__ inline float fast_tanh(float x) { return 1.0f - 2.0f * __frcp_rn(__expf(2.0f * x) + 1.0f); }

// ---------------------------------------------------------------------------
// WMMA fragment loaders.
// A (16x32 bf16): lanes hold M=0..15 / K=16*grp..16*grp+15 (contiguous);
// grp = lane>>4.  Two 16B LDS reads per fragment.
// B (32x16 bf16): pre-packed so each lane's 16 halves are contiguous.
// ---------------------------------------------------------------------------
__device__ inline v16bf load_afrag_lds(const __bf16* base, int strideHalves,
                                       int lane, int mtile, int ktile) {
    int m   = (lane & 15) + mtile * 16;
    int grp = lane >> 4;
    const __bf16* p = base + m * strideHalves + ktile * 32 + grp * 16;
    union { v16bf v; float4 q[2]; } u;
    u.q[0] = *reinterpret_cast<const float4*>(p);        // halves 0..7
    u.q[1] = *reinterpret_cast<const float4*>(p + 8);    // halves 8..15
    return u.v;
}
__device__ inline v16bf load_bfrag_ptr(const __bf16* p) {
    union { v16bf v; float4 q[2]; } u;
    u.q[0] = *reinterpret_cast<const float4*>(p);
    u.q[1] = *reinterpret_cast<const float4*>(p + 8);
    return u.v;
}
#define WMMA_BF16(A, B, C) \
    __builtin_amdgcn_wmma_f32_16x16x32_bf16(false, (A), false, (B), (short)0, (C), false, false)

// ---------------------------------------------------------------------------
// Prep kernel 1: build A_hat CSR (chain => <=3 nnz/row) + concatenated bias
// ---------------------------------------------------------------------------
__global__ void build_misc_kernel(const int* __restrict__ ei,
                                  const float* __restrict__ bi, const float* __restrict__ bf_,
                                  const float* __restrict__ bo, const float* __restrict__ bg,
                                  float* __restrict__ ball,
                                  int* __restrict__ nnz, int* __restrict__ colA,
                                  float* __restrict__ valA) {
    int tid = threadIdx.x;
    if (tid < HDIM) {
        ball[tid]            = bi[tid];
        ball[HDIM + tid]     = bf_[tid];
        ball[2 * HDIM + tid] = bo[tid];
        ball[3 * HDIM + tid] = bg[tid];
    }
    if (tid == 0) {
        float A[32][32];
        float deg[32], dinv[32];
        for (int m = 0; m < 32; ++m)
            for (int n = 0; n < 32; ++n) A[m][n] = 0.0f;
        for (int n = 0; n < 32; ++n) deg[n] = 0.0f;
        for (int n = 0; n < NNODES; ++n) deg[n] = 1.0f;            // self loops
        for (int e = 0; e < NEDGES; ++e) deg[ei[NEDGES + e]] += 1.0f;
        for (int n = 0; n < NNODES; ++n) dinv[n] = rsqrtf(deg[n]);
        for (int e = 0; e < NEDGES; ++e) {
            int s = ei[e], d = ei[NEDGES + e];
            A[d][s] += dinv[s] * dinv[d];
        }
        for (int n = 0; n < NNODES; ++n) A[n][n] += dinv[n] * dinv[n];
        for (int m = 0; m < 32; ++m) {
            int c = 0;
            if (m < NNODES)
                for (int n = 0; n < NNODES; ++n)
                    if (A[m][n] != 0.0f && c < 8) {
                        colA[m * 8 + c] = n;
                        valA[m * 8 + c] = A[m][n];
                        ++c;
                    }
            nnz[m] = c;
        }
    }
}

// ---------------------------------------------------------------------------
// Prep kernel 2: pack [W_i|W_f|W_o|W_g] ([262,256] each) into bf16 WMMA
// B-fragment layout for the [288, 1024] virtual matrix.
// ---------------------------------------------------------------------------
__global__ void pack_gates_kernel(const float* __restrict__ Wi, const float* __restrict__ Wf,
                                  const float* __restrict__ Wo, const float* __restrict__ Wg,
                                  __bf16* __restrict__ dst) {
    int idx = blockIdx.x * 256 + threadIdx.x;
    const int total = KT_G * NT_G * 32 * 16;
    if (idx >= total) return;
    int hh   = idx & 15;
    int lane = (idx >> 4) & 31;
    int nt   = (idx >> 9) & 63;
    int kt   = idx >> 15;
    int k    = kt * 32 + ((lane >> 4) << 4) + hh;   // K index in [0,288)
    int gcol = nt * 16 + (lane & 15);               // column in [0,1024)
    int gate = gcol >> 8;
    int j    = gcol & 255;
    float v = 0.0f;
    if (k < D_COMB) {
        const float* W = (gate == 0) ? Wi : (gate == 1) ? Wf : (gate == 2) ? Wo : Wg;
        v = W[k * HDIM + j];
    }
    dst[idx] = f2bf(v);
}

// ---------------------------------------------------------------------------
// Prep kernel 3: generic pack of one [K, Ncols] fp32 matrix -> bf16 B layout
// ---------------------------------------------------------------------------
__global__ void pack_one_kernel(const float* __restrict__ W, __bf16* __restrict__ dst,
                                int K, int Ncols, int NT, int total) {
    int idx = blockIdx.x * 256 + threadIdx.x;
    if (idx >= total) return;
    int hh   = idx & 15;
    int lane = (idx >> 4) & 31;
    int nt   = (idx >> 9) % NT;
    int kt   = idx / (512 * NT);
    int k    = kt * 32 + ((lane >> 4) << 4) + hh;
    int col  = nt * 16 + (lane & 15);
    float v  = (k < K && col < Ncols) ? W[k * Ncols + col] : 0.0f;
    dst[idx] = f2bf(v);
}

// ---------------------------------------------------------------------------
// Main fused kernel: one workgroup (16 waves) per batch element.
//   - h[32,256] bf16 resident in LDS for all 1024 steps
//   - c kept in WMMA accumulator-fragment registers (2 v8f per wave)
//   - wave w owns hidden columns [16w,16w+16) for ALL four gates
//   - rolled k-loop (unroll 1): 4 ds_load_b128 + 8 global_load_b128 +
//     8 v_wmma per iteration, no spills
//   - straight-line activations (no EXEC-divergent libm tanh)
// ---------------------------------------------------------------------------
__global__ __launch_bounds__(NTHREADS)
void gclstm_kernel(const float* __restrict__ x,
                   const int* __restrict__ g_nnz, const int* __restrict__ g_col,
                   const float* __restrict__ g_val, const float* __restrict__ ball,
                   const __bf16* __restrict__ Wg_pk, const __bf16* __restrict__ Wm_pk,
                   const __bf16* __restrict__ Wf_pk,
                   const float* __restrict__ bm, const float* __restrict__ bfc,
                   const int* __restrict__ ei,
                   float* __restrict__ out) {
    __shared__ __align__(16) __bf16 sH[32 * HDIM];     // hidden state, bf16
    __shared__ __align__(16) __bf16 sAgg[32 * KPAD];   // agg (step) / edge_x (edge)
    __shared__ __align__(16) __bf16 sMid[16 * HDIM];   // edge-phase mlp1 output
    __shared__ int   s_nnz[32];
    __shared__ int   s_col[32 * 8];
    __shared__ float s_val[32 * 8];
    __shared__ float s_x[NNODES * D_IN];

    const int tid  = threadIdx.x;
    const int b    = blockIdx.x;
    const int wave = tid >> 5;          // 0..15
    const int lane = tid & 31;
    const int grp  = lane >> 4;
    const int ln16 = lane & 15;

    // ---- init LDS -----------------------------------------------------
    for (int i = tid; i < 32 * HDIM; i += NTHREADS) sH[i] = f2bf(0.0f);
    for (int i = tid; i < 32 * KPAD; i += NTHREADS) sAgg[i] = f2bf(0.0f);
    if (tid < 32)      s_nnz[tid] = g_nnz[tid];
    if (tid < 32 * 8) { s_col[tid] = g_col[tid]; s_val[tid] = g_val[tid]; }

    // ---- per-lane gate biases (column-resident, loop-invariant) --------
    const int jbase = wave * 16;        // 16-wide hidden slice per wave
    float bgate[4];
#pragma unroll
    for (int g = 0; g < 4; ++g) bgate[g] = ball[g * HDIM + jbase + ln16];

    // per-wave/lane weight base: gate 0, n-tile = wave, this lane
    const __bf16* wg_lane = Wg_pk + ((wave * 32 + lane) << 4);

    // ---- cell state in D-fragment layout (2 m-tiles x 16 cols) ---------
    const v8f vzero = {0.f, 0.f, 0.f, 0.f, 0.f, 0.f, 0.f, 0.f};
    v8f cst[2];
    cst[0] = vzero; cst[1] = vzero;

    __syncthreads();

    // ===================== recurrent time loop ==========================
    for (int t = 0; t < TSTEPS; ++t) {
        // stage x_t into LDS
        const float* xt = x + (((size_t)b * TSTEPS + t) * NNODES) * D_IN;
        if (tid < NNODES * D_IN) s_x[tid] = xt[tid];
        __syncthreads();   // also fences previous step's sH writes / sAgg reads

        // sparse aggregation: agg[m,k] = sum_n A[m,n] * comb[n,k]
        for (int idx = tid; idx < NNODES * D_COMB; idx += NTHREADS) {
            int m = idx / D_COMB;
            int k = idx - m * D_COMB;
            float s = 0.0f;
            int cnt = s_nnz[m];
            for (int q = 0; q < cnt; ++q) {
                int   n = s_col[m * 8 + q];
                float a = s_val[m * 8 + q];
                float cv = (k < D_IN) ? s_x[n * D_IN + k]
                                      : bf2f(sH[n * HDIM + (k - D_IN)]);
                s += a * cv;
            }
            sAgg[m * KPAD + k] = f2bf(s);
        }
        __syncthreads();

        // GEMM: wave w -> columns {g*256 + 16w .. +16}, gates g=0..3
        v8f acc[4][2];
#pragma unroll
        for (int g = 0; g < 4; ++g) { acc[g][0] = vzero; acc[g][1] = vzero; }

        const __bf16* bp = wg_lane;
#pragma unroll 1           // keep rolled: avoids whole-loop load hoist + spill
        for (int kt = 0; kt < KT_G; ++kt) {
            v16bf a0 = load_afrag_lds(sAgg, KPAD, lane, 0, kt);
            v16bf a1 = load_afrag_lds(sAgg, KPAD, lane, 1, kt);
            __builtin_prefetch(bp + KSLAB_G, 0, 1);   // warm next k-slab (L2)
#pragma unroll
            for (int g = 0; g < 4; ++g) {
                v16bf bb = load_bfrag_ptr(bp + g * GATE_OFF);
                acc[g][0] = WMMA_BF16(a0, bb, acc[g][0]);
                acc[g][1] = WMMA_BF16(a1, bb, acc[g][1]);
            }
            bp += KSLAB_G;
        }

        // LSTM pointwise update, c in registers, h -> LDS (bf16)
#pragma unroll
        for (int mt = 0; mt < 2; ++mt) {
#pragma unroll
            for (int v = 0; v < 8; ++v) {
                float iv = acc[0][mt][v] + bgate[0];
                float fv = acc[1][mt][v] + bgate[1];
                float ov = acc[2][mt][v] + bgate[2];
                float gv = acc[3][mt][v] + bgate[3];
                float c0 = cst[mt][v];
                float cn = sigm(fv) * c0 + sigm(iv) * fast_tanh(gv);
                cst[mt][v] = cn;
                float hv = sigm(ov) * fast_tanh(cn);
                int m = mt * 16 + v + (grp << 3);
                if (m < NNODES) {
                    int j = jbase + ln16;
                    sH[m * HDIM + j] = f2bf(hv);
                }
            }
        }
        // next iteration's barrier (after x staging) fences sH before reuse
    }
    __syncthreads();

    // ===================== edge MLP head ================================
    __bf16* eX = sAgg;   // reuse: 16 x 512 bf16 fits in 32 x 288 buffer
    for (int ec = 0; ec < 4; ++ec) {
        // build edge_x chunk: [h[dst] | h[src]] for edges ec*16 .. +16
        for (int idx = tid; idx < 16 * 512; idx += NTHREADS) {
            int r = idx >> 9;
            int k = idx & 511;
            int e = ec * 16 + r;
            __bf16 v = f2bf(0.0f);
            if (e < NEDGES) {
                int node = (k < HDIM) ? ei[NEDGES + e] : ei[e];  // dst then src
                v = sH[node * HDIM + (k & 255)];
            }
            eX[r * 512 + k] = v;
        }
        __syncthreads();

        // mlp1: [16,512] @ [512,256], wave w -> n-tile w
        v8f a2 = vzero;
        const __bf16* mp = Wm_pk + ((wave * 32 + lane) << 4);
#pragma unroll 1
        for (int kt = 0; kt < KT_M; ++kt) {
            v16bf a0 = load_afrag_lds(eX, 512, lane, 0, kt);
            v16bf bb = load_bfrag_ptr(mp);
            a2 = WMMA_BF16(a0, bb, a2);
            mp += NT_M * FRAG_HALVES;
        }
#pragma unroll
        for (int v = 0; v < 8; ++v) {
            int col = wave * 16 + ln16;
            float vv = a2[v] + bm[col];
            vv = vv > 0.0f ? vv : 0.0f;                           // ReLU
            int m = v + (grp << 3);
            sMid[m * HDIM + col] = f2bf(vv);
        }
        __syncthreads();

        // fc: [16,256] @ [256,32], waves 0..1 each take one 16-col tile
        if (wave < 2) {
            v8f a3 = vzero;
            const __bf16* fp = Wf_pk + ((wave * 32 + lane) << 4);
#pragma unroll 1
            for (int kt = 0; kt < KT_F; ++kt) {
                v16bf a0 = load_afrag_lds(sMid, HDIM, lane, 0, kt);
                v16bf bb = load_bfrag_ptr(fp);
                a3 = WMMA_BF16(a0, bb, a3);
                fp += NT_F * FRAG_HALVES;
            }
#pragma unroll
            for (int v = 0; v < 8; ++v) {
                int m = v + (grp << 3);
                int e = ec * 16 + m;
                int col = wave * 16 + ln16;
                if (e < NEDGES)
                    out[((size_t)b * NEDGES + e) * D_OUT + col] = a3[v] + bfc[col];
            }
        }
        __syncthreads();
    }
}

// ---------------------------------------------------------------------------
// Host launcher
// ---------------------------------------------------------------------------
extern "C" void kernel_launch(void* const* d_in, const int* in_sizes, int n_in,
                              void* d_out, int out_size, void* d_ws, size_t ws_size,
                              hipStream_t stream) {
    const float* x    = (const float*)d_in[0];
    const float* Wi   = (const float*)d_in[1];
    const float* bi   = (const float*)d_in[2];
    const float* Wf   = (const float*)d_in[3];
    const float* bf_  = (const float*)d_in[4];
    const float* Wo   = (const float*)d_in[5];
    const float* bo   = (const float*)d_in[6];
    const float* Wg   = (const float*)d_in[7];
    const float* bg   = (const float*)d_in[8];
    const float* Wm1  = (const float*)d_in[9];
    const float* bm1  = (const float*)d_in[10];
    const float* Wfc  = (const float*)d_in[11];
    const float* bfc  = (const float*)d_in[12];
    const int*   ei   = (const int*)d_in[13];
    float*       out  = (float*)d_out;

    // workspace layout (bytes)
    char* ws = (char*)d_ws;
    int*    ws_nnz  = (int*)(ws + 0);            //   128 B
    int*    ws_col  = (int*)(ws + 128);          //  1024 B
    float*  ws_val  = (float*)(ws + 1152);       //  1024 B
    float*  ws_ball = (float*)(ws + 2176);       //  4096 B
    __bf16* ws_Wg   = (__bf16*)(ws + 6400);      // 9*64*512*2  = 589824 B
    __bf16* ws_Wm   = (__bf16*)(ws + 596224);    // 16*16*512*2 = 262144 B
    __bf16* ws_Wf   = (__bf16*)(ws + 858368);    // 8*2*512*2   =  16384 B
    (void)ws_size; (void)in_sizes; (void)n_in; (void)out_size;

    build_misc_kernel<<<1, 256, 0, stream>>>(ei, bi, bf_, bo, bg,
                                             ws_ball, ws_nnz, ws_col, ws_val);

    const int totG = KT_G * NT_G * 512;
    pack_gates_kernel<<<(totG + 255) / 256, 256, 0, stream>>>(Wi, Wf, Wo, Wg, ws_Wg);

    const int totM = KT_M * NT_M * 512;
    pack_one_kernel<<<(totM + 255) / 256, 256, 0, stream>>>(Wm1, ws_Wm, 512, HDIM, NT_M, totM);

    const int totF = KT_F * NT_F * 512;
    pack_one_kernel<<<(totF + 255) / 256, 256, 0, stream>>>(Wfc, ws_Wf, HDIM, D_OUT, NT_F, totF);

    gclstm_kernel<<<BATCH, NTHREADS, 0, stream>>>(x, ws_nnz, ws_col, ws_val, ws_ball,
                                                  ws_Wg, ws_Wm, ws_Wf, bm1, bfc, ei, out);
}
